// CausalSelfAttention_39900246180529
// MI455X (gfx1250) — compile-verified, matrix operand order checked by reading, not run on hardware
//
#include <hip/hip_runtime.h>
#include <hip/hip_bf16.h>

// ---------------------------------------------------------------------------
// CDNA5 (gfx1250) causal self-attention, bf16 WMMA pipeline.
// B=4, T=2048, C=1024, NH=16, HD=64.
//   1) f32 -> bf16 converts (x, w_qkv, w_proj)
//   2) qkv = x @ w_qkv          (WMMA bf16, dbl-buffered LDS, async-to-LDS)
//   3) flash attention          (WMMA QK^T + PV, online softmax)
//   4) out = attn @ w_proj      (WMMA bf16, f32 out)
// ---------------------------------------------------------------------------

typedef __attribute__((ext_vector_type(16))) __bf16 v16bf;
typedef __attribute__((ext_vector_type(8)))  __bf16 v8bf;
typedef __attribute__((ext_vector_type(2)))  __bf16 v2bf;
typedef __attribute__((ext_vector_type(8)))  float  v8f;
typedef __attribute__((ext_vector_type(4)))  int    v4i;

#define LANE (threadIdx.x & 31)

// --- CDNA5 async global->LDS copy (ASYNCcnt path), with safe fallback -------
#if defined(__gfx1250__) && __has_builtin(__builtin_amdgcn_global_load_async_to_lds_b128)
#define HAVE_ASYNC_LDS 1
#define AS1 __attribute__((address_space(1)))
#define AS3 __attribute__((address_space(3)))
__device__ __forceinline__ void async_copy16(const __bf16* g, __bf16* l) {
    __builtin_amdgcn_global_load_async_to_lds_b128((AS1 v4i*)(void*)g,
                                                   (AS3 v4i*)(void*)l, 0, 0);
}
#define ASYNC_WAIT() asm volatile("s_wait_asynccnt 0x0" ::: "memory")
#else
#define HAVE_ASYNC_LDS 0
__device__ __forceinline__ void async_copy16(const __bf16* g, __bf16* l) {
    *(v8bf*)l = *(const v8bf*)g;
}
#define ASYNC_WAIT()
#endif

// Load a 16x32 (A) or 32x16-as-[n][k] (B) bf16 WMMA fragment from LDS.
// Per 7.12.2: lane l holds row (row0 + (l&15)); its 16 elements are two
// contiguous K-runs of 8: k = k0 + 8*(l>>4) + {0..7} and +16 + {0..7}.
__device__ __forceinline__ v16bf lds_frag(const __bf16* tile, int row0, int ld, int k0) {
    const int l = LANE;
    const __bf16* p = tile + (row0 + (l & 15)) * ld + k0 + ((l >> 4) << 3);
    v8bf lo = *(const v8bf*)p;
    v8bf hi = *(const v8bf*)(p + 16);
    return __builtin_shufflevector(lo, hi, 0,1,2,3,4,5,6,7,8,9,10,11,12,13,14,15);
}

__device__ __forceinline__ v8f wmma_bf16(v16bf a, v16bf b, v8f c) {
    return __builtin_amdgcn_wmma_f32_16x16x32_bf16(false, a, false, b, (short)0, c, false, false);
}

// ---------------------------------------------------------------------------
// f32 -> bf16 elementwise convert (8 elements / thread)
// ---------------------------------------------------------------------------
__global__ __launch_bounds__(256) void f32_to_bf16_kernel(const float* __restrict__ in,
                                                          __bf16* __restrict__ out, int n8) {
    int idx = blockIdx.x * 256 + threadIdx.x;
    if (idx >= n8) return;
    v8f a = *((const v8f*)in + idx);
    *((v8bf*)out + idx) = __builtin_convertvector(a, v8bf);
}

// ---------------------------------------------------------------------------
// Tiled bf16 GEMM: C[M,N] = A[M,K] @ B[K,N]. Block = 128x256x32, 256 threads
// (8 waves as 2x4); each wave computes a 64x64 sub-tile = 4x4 WMMA tiles
// (16 WMMAs per 8 fragment loads). Double-buffered LDS; A fills use the
// async-to-LDS path so next-tile fills overlap current-tile WMMAs.
// ---------------------------------------------------------------------------
template <bool STORE_BF16>
__global__ __launch_bounds__(256) void gemm_bf16_wmma(const __bf16* __restrict__ A,
                                                      const __bf16* __restrict__ Bm,
                                                      void* __restrict__ Cv,
                                                      int M, int N, int K) {
    constexpr int BM = 128, BN = 256, BK = 32, LDT = BK + 8;  // pad keeps 16B align
    __shared__ __align__(16) __bf16 sA[2][BM * LDT];          // [m][k]
    __shared__ __align__(16) __bf16 sB[2][BN * LDT];          // transposed: [n][k]

    const int t    = threadIdx.x;
    const int wave = t >> 5;
    const int lane = t & 31;
    const int mb   = (int)blockIdx.y * BM;
    const int nb   = (int)blockIdx.x * BN;
    const int wm   = (wave & 1) * 64;    // 2 waves along M
    const int wn   = (wave >> 1) * 64;   // 4 waves along N

    v8f acc[4][4];
#pragma unroll
    for (int i = 0; i < 4; ++i)
#pragma unroll
        for (int j = 0; j < 4; ++j) acc[i][j] = (v8f){0.f,0.f,0.f,0.f,0.f,0.f,0.f,0.f};

    auto fill = [&](int buf, int k0) {
        // A tile: 128x32 row-major, 16B per lane via async-to-LDS.
#pragma unroll
        for (int i = 0; i < 2; ++i) {
            int row = i * 64 + (t >> 2);
            int kk  = (t & 3) * 8;
            async_copy16(&A[(size_t)(mb + row) * K + k0 + kk], &sA[buf][row * LDT + kk]);
        }
        // B tile transposed into [n][k]: read two adjacent k-rows, store
        // bf16 pairs as b32 (k-pairs contiguous in the [n][k] layout).
#pragma unroll
        for (int i = 0; i < 2; ++i) {
            int idx = i * 256 + t;
            int kr  = (idx >> 5) * 2;       // 0..30
            int ns  = (idx & 31) * 8;       // 0..248
            v8bf w0 = *(const v8bf*)&Bm[(size_t)(k0 + kr) * N + nb + ns];
            v8bf w1 = *(const v8bf*)&Bm[(size_t)(k0 + kr + 1) * N + nb + ns];
#pragma unroll
            for (int j = 0; j < 8; ++j) {
                v2bf p; p.x = w0[j]; p.y = w1[j];
                *(v2bf*)&sB[buf][(ns + j) * LDT + kr] = p;
            }
        }
    };

    fill(0, 0);
    ASYNC_WAIT();
    __syncthreads();

    int cur = 0;
    for (int k0 = 0; k0 < K; k0 += BK) {
        if (k0 + BK < K) fill(cur ^ 1, k0 + BK);   // overlap with WMMAs below

        v16bf af[4];
#pragma unroll
        for (int i = 0; i < 4; ++i) af[i] = lds_frag(sA[cur], wm + i * 16, LDT, 0);
#pragma unroll
        for (int j = 0; j < 4; ++j) {
            v16bf bfj = lds_frag(sB[cur], wn + j * 16, LDT, 0);
#pragma unroll
            for (int i = 0; i < 4; ++i) acc[i][j] = wmma_bf16(af[i], bfj, acc[i][j]);
        }
        ASYNC_WAIT();
        __syncthreads();
        cur ^= 1;
    }

    // Epilogue: D layout (7.12.2): elem r -> row = r + 8*(lane>>4), col = lane&15.
#pragma unroll
    for (int i = 0; i < 4; ++i)
#pragma unroll
        for (int j = 0; j < 4; ++j)
#pragma unroll
            for (int r = 0; r < 8; ++r) {
                int m = mb + wm + i * 16 + r + ((lane >> 4) << 3);
                int n = nb + wn + j * 16 + (lane & 15);
                if (STORE_BF16)
                    ((__bf16*)Cv)[(size_t)m * N + n] = (__bf16)acc[i][j][r];
                else
                    ((float*)Cv)[(size_t)m * N + n] = acc[i][j][r];
            }
}

// ---------------------------------------------------------------------------
// Flash attention: one block = 64 queries for one (b, h). 128 threads =
// 4 waves; wave w owns queries [16w, 16w+16). Online softmax, causal.
// qkv layout: [B*T, 3C] bf16; q at col h*64, k at 1024+h*64, v at 2048+h*64.
// ---------------------------------------------------------------------------
__global__ __launch_bounds__(128) void flash_attn_kernel(const __bf16* __restrict__ qkv,
                                                         __bf16* __restrict__ outb) {
    constexpr int LDH = 72;  // 64 + 8 pad
    __shared__ __align__(16) __bf16 sQ[64 * LDH];
    __shared__ __align__(16) __bf16 sK[64 * LDH];       // [key][hd]
    __shared__ __align__(16) __bf16 sVT[64 * LDH];      // [hd][key]
    __shared__ __align__(16) __bf16 sP[4][16 * LDH];    // per-wave P tile [m][key]

    const int t = threadIdx.x, lane = t & 31, wave = t >> 5;
    const int qt = blockIdx.x, h = blockIdx.y, b = blockIdx.z;
    const int qbase = qt * 64;
    const int m0 = wave * 16;
    const __bf16* base = qkv + (size_t)(b * 2048) * 3072 + h * 64;

    // Load Q tile (64x64) once (async; completed by the wait in the kv loop).
#pragma unroll
    for (int i = 0; i < 4; ++i) {
        int lin = i * 128 + t;
        int row = lin >> 3, ds = (lin & 7) * 8;
        async_copy16(&base[(size_t)(qbase + row) * 3072 + ds], &sQ[row * LDH + ds]);
    }

    float mi[8], li[8];
    v8f o[4];
#pragma unroll
    for (int r = 0; r < 8; ++r) { mi[r] = -INFINITY; li[r] = 0.f; }
#pragma unroll
    for (int d = 0; d < 4; ++d) o[d] = (v8f){0.f,0.f,0.f,0.f,0.f,0.f,0.f,0.f};
    const float scale = 0.125f;  // 1/sqrt(64)

    for (int jt = 0; jt <= qt; ++jt) {
        const int kbase = jt * 64;
        // K tile (row-major) via async copies.
#pragma unroll
        for (int i = 0; i < 4; ++i) {
            int lin = i * 128 + t;
            int key = lin >> 3, ds = (lin & 7) * 8;
            async_copy16(&base[(size_t)(kbase + key) * 3072 + 1024 + ds], &sK[key * LDH + ds]);
        }
        // V tile transposed [hd][key]: two adjacent keys packed as b32 pairs.
#pragma unroll
        for (int i = 0; i < 2; ++i) {
            int idx = i * 128 + t;
            int kp  = (idx >> 3) * 2;       // key pair 0..62
            int ds  = (idx & 7) * 8;        // 0..56
            v8bf v0 = *(const v8bf*)&base[(size_t)(kbase + kp) * 3072 + 2048 + ds];
            v8bf v1 = *(const v8bf*)&base[(size_t)(kbase + kp + 1) * 3072 + 2048 + ds];
#pragma unroll
            for (int j = 0; j < 8; ++j) {
                v2bf p; p.x = v0[j]; p.y = v1[j];
                *(v2bf*)&sVT[(ds + j) * LDH + kp] = p;
            }
        }
        ASYNC_WAIT();
        __syncthreads();

        // S = Q K^T : 4 tiles of 16x16, each 2 chained WMMAs over HD=64.
        v16bf aq0 = lds_frag(sQ, m0, LDH, 0);
        v16bf aq1 = lds_frag(sQ, m0, LDH, 32);
        v8f s[4];
#pragma unroll
        for (int jn = 0; jn < 4; ++jn) {
            v16bf bk0 = lds_frag(sK, jn * 16, LDH, 0);
            v16bf bk1 = lds_frag(sK, jn * 16, LDH, 32);
            v8f z = (v8f){0.f,0.f,0.f,0.f,0.f,0.f,0.f,0.f};
            z = wmma_bf16(aq0, bk0, z);
            s[jn] = wmma_bf16(aq1, bk1, z);
        }

        // Scale + causal mask (diagonal tile only; block-local q/k compare).
        float rmax[8];
#pragma unroll
        for (int r = 0; r < 8; ++r) rmax[r] = -INFINITY;
#pragma unroll
        for (int jn = 0; jn < 4; ++jn)
#pragma unroll
            for (int r = 0; r < 8; ++r) {
                float v = s[jn][r] * scale;
                if (jt == qt) {
                    int q = m0 + r + ((lane >> 4) << 3);
                    int k = jn * 16 + (lane & 15);
                    if (k > q) v = -INFINITY;
                }
                s[jn][r] = v;
                rmax[r] = fmaxf(rmax[r], v);
            }
        // Row reductions across the 16 lanes that share a row (wave32).
#pragma unroll
        for (int r = 0; r < 8; ++r) {
            rmax[r] = fmaxf(rmax[r], __shfl_xor(rmax[r], 1, 32));
            rmax[r] = fmaxf(rmax[r], __shfl_xor(rmax[r], 2, 32));
            rmax[r] = fmaxf(rmax[r], __shfl_xor(rmax[r], 4, 32));
            rmax[r] = fmaxf(rmax[r], __shfl_xor(rmax[r], 8, 32));
        }
        float alpha[8], rsum[8];
#pragma unroll
        for (int r = 0; r < 8; ++r) {
            float mnew = fmaxf(mi[r], rmax[r]);
            alpha[r] = __expf(mi[r] - mnew);
            mi[r] = mnew;
            rsum[r] = 0.f;
        }
#pragma unroll
        for (int jn = 0; jn < 4; ++jn)
#pragma unroll
            for (int r = 0; r < 8; ++r) {
                float p = __expf(s[jn][r] - mi[r]);
                s[jn][r] = p;
                rsum[r] += p;
            }
#pragma unroll
        for (int r = 0; r < 8; ++r) {
            rsum[r] += __shfl_xor(rsum[r], 1, 32);
            rsum[r] += __shfl_xor(rsum[r], 2, 32);
            rsum[r] += __shfl_xor(rsum[r], 4, 32);
            rsum[r] += __shfl_xor(rsum[r], 8, 32);
            li[r] = li[r] * alpha[r] + rsum[r];
        }
#pragma unroll
        for (int d = 0; d < 4; ++d)
#pragma unroll
            for (int r = 0; r < 8; ++r) o[d][r] *= alpha[r];

        // P (D-layout) -> bf16 -> per-wave LDS -> reload in A-layout.
        // Per-wave private region; LDS ops are in-order within a wave.
#pragma unroll
        for (int jn = 0; jn < 4; ++jn)
#pragma unroll
            for (int r = 0; r < 8; ++r)
                sP[wave][(r + ((lane >> 4) << 3)) * LDH + jn * 16 + (lane & 15)] =
                    (__bf16)s[jn][r];

        v16bf ap0 = lds_frag(sP[wave], 0, LDH, 0);
        v16bf ap1 = lds_frag(sP[wave], 0, LDH, 32);
#pragma unroll
        for (int d = 0; d < 4; ++d) {
            v16bf bv0 = lds_frag(sVT, d * 16, LDH, 0);
            v16bf bv1 = lds_frag(sVT, d * 16, LDH, 32);
            o[d] = wmma_bf16(ap0, bv0, o[d]);
            o[d] = wmma_bf16(ap1, bv1, o[d]);
        }
        __syncthreads();
    }

    // Normalize and store to attn buffer [B*T, C] at col h*64.
    __bf16* op = outb + (size_t)(b * 2048 + qbase + m0) * 1024 + h * 64;
#pragma unroll
    for (int d = 0; d < 4; ++d)
#pragma unroll
        for (int r = 0; r < 8; ++r)
            op[(size_t)(r + ((lane >> 4) << 3)) * 1024 + d * 16 + (lane & 15)] =
                (__bf16)(o[d][r] / li[r]);
}

// ---------------------------------------------------------------------------
// Host launcher
// ---------------------------------------------------------------------------
extern "C" void kernel_launch(void* const* d_in, const int* in_sizes, int n_in,
                              void* d_out, int out_size, void* d_ws, size_t ws_size,
                              hipStream_t stream) {
    const float* x     = (const float*)d_in[0];   // [4,2048,1024]
    const float* wqkv  = (const float*)d_in[1];   // [1024,3072]
    const float* wproj = (const float*)d_in[2];   // [1024,1024]
    float* out = (float*)d_out;                    // [4,2048,1024]

    char* ws = (char*)d_ws;
    __bf16* x16   = (__bf16*)(ws);                           // 16 MB
    __bf16* wq16  = (__bf16*)(ws + 16777216);                //  6 MB
    __bf16* wp16  = (__bf16*)(ws + 23068672);                //  2 MB
    __bf16* qkv16 = (__bf16*)(ws + 25165824);                // 48 MB
    __bf16* att16 = (__bf16*)(ws + 75497472);                // 16 MB

    // Converts
    f32_to_bf16_kernel<<<(8388608 / 8) / 256, 256, 0, stream>>>(x, x16, 8388608 / 8);
    f32_to_bf16_kernel<<<(3145728 / 8) / 256, 256, 0, stream>>>(wqkv, wq16, 3145728 / 8);
    f32_to_bf16_kernel<<<(1048576 / 8) / 256, 256, 0, stream>>>(wproj, wp16, 1048576 / 8);

    // qkv = x16 @ wq16  : [8192,1024] x [1024,3072] -> bf16 [8192,3072]
    gemm_bf16_wmma<true><<<dim3(3072 / 256, 8192 / 128), 256, 0, stream>>>(
        x16, wq16, (void*)qkv16, 8192, 3072, 1024);

    // flash attention -> att16 [8192,1024] bf16
    flash_attn_kernel<<<dim3(32, 16, 4), 128, 0, stream>>>(qkv16, att16);

    // out = att16 @ wp16 : [8192,1024] x [1024,1024] -> f32
    gemm_bf16_wmma<false><<<dim3(1024 / 256, 8192 / 128), 256, 0, stream>>>(
        att16, wp16, (void*)out, 8192, 1024, 1024);
}
